// MultiHeadAttentionBlock_33672543601180
// MI455X (gfx1250) — matrix operand order repaired and checked by reading, not compiled, not run on hardware
//
#include <hip/hip_runtime.h>

// ---------------------------------------------------------------------------
// MI455X (gfx1250) multi-head attention block:
// bf16 WMMA GEMMs with TDM (tensor_load_to_lds) double-buffered tiles,
// flash attention with online softmax.
// ---------------------------------------------------------------------------

typedef __bf16 bf16_t;
typedef __attribute__((ext_vector_type(8)))  __bf16 bf16x8;
typedef __attribute__((ext_vector_type(16))) __bf16 v16bf;
typedef __attribute__((ext_vector_type(8)))  float  v8f;
typedef __attribute__((ext_vector_type(4)))  unsigned int u32x4;
typedef __attribute__((ext_vector_type(8)))  int i32x8;
typedef __attribute__((ext_vector_type(4)))  int i32x4;

constexpr int kB  = 4;
constexpr int kS  = 2048;
constexpr int kD  = 1024;
constexpr int kH  = 16;
constexpr int kDH = 64;
constexpr int kM  = kB * kS;      // 8192 rows

#if defined(__has_builtin)
#  if __has_builtin(__builtin_amdgcn_tensor_load_to_lds)
#    define HAVE_TDM 1
#  endif
#endif
#ifndef HAVE_TDM
#  define HAVE_TDM 0
#endif

__device__ __forceinline__ bf16_t to_bf16(float f) {
    unsigned u = __builtin_bit_cast(unsigned, f);
    u += 0x7fffu + ((u >> 16) & 1u);          // round-to-nearest-even
    unsigned short s = (unsigned short)(u >> 16);
    return __builtin_bit_cast(bf16_t, s);
}

__device__ __forceinline__ v16bf combine16(bf16x8 lo, bf16x8 hi) {
    return __builtin_shufflevector(lo, hi, 0,1,2,3,4,5,6,7,8,9,10,11,12,13,14,15);
}

__device__ __forceinline__ v8f wmma_bf16(v16bf a, v16bf b, v8f c) {
    return __builtin_amdgcn_wmma_f32_16x16x32_bf16(false, a, false, b, (short)0, c, false, false);
}

#if HAVE_TDM
// Build a 2D tensor-DMA descriptor (D#) per CDNA5 ISA
// and issue TENSOR_LOAD_TO_LDS. 2-byte elements; groups 2/3 zero (2D tensor).
// This toolchain exposes the 6-arg builtin: (g0 u32x4, g1 i32x8, g2 i32x4,
// g3 i32x4, extra i32x8, cpol i32).
__device__ __forceinline__ void tdm_load_2d(unsigned lds_off, const void* gptr,
                                            unsigned tensor_d0, unsigned tensor_d1,
                                            unsigned stride0_el,
                                            unsigned tile_d0, unsigned tile_d1) {
    const unsigned long long ga = (unsigned long long)gptr;
    u32x4 g0;
    g0[0] = 1u;                                               // count=1, user desc
    g0[1] = lds_off;                                          // lds_addr (bytes)
    g0[2] = (unsigned)(ga & 0xFFFFFFFFu);                     // global_addr[31:0]
    g0[3] = (unsigned)((ga >> 32) & 0x01FFFFFFu) | (2u << 30);// addr[56:32] | type=2
    i32x8 g1;
    g1[0] = (int)(1u << 16);                                  // data_size=1 (2B), wg_mask=0
    g1[1] = (int)((tensor_d0 & 0xFFFFu) << 16);               // dim0[15:0] (<<16)
    g1[2] = (int)(((tensor_d0 >> 16) & 0xFFFFu) |
                  ((tensor_d1 & 0xFFFFu) << 16));             // dim0[31:16] | dim1[15:0]
    g1[3] = (int)(((tensor_d1 >> 16) & 0xFFFFu) |
                  ((tile_d0 & 0xFFFFu) << 16));               // dim1[31:16] | tile_dim0
    g1[4] = (int)(tile_d1 & 0xFFFFu);                         // tile_dim1, tile_dim2=0
    g1[5] = (int)stride0_el;                                  // tensor_dim0_stride[31:0]
    g1[6] = 0;                                                // stride0 hi | dim1_stride lo
    g1[7] = 0;
    i32x4 z4 = {0, 0, 0, 0};
    i32x8 z8 = {0, 0, 0, 0, 0, 0, 0, 0};
    __builtin_amdgcn_tensor_load_to_lds(g0, g1, z4, z4, z8, 0);
}
#endif

// ---------------------------------------------------------------------------
// fp32 -> bf16 conversion
// ---------------------------------------------------------------------------
__global__ void cvt_f32_bf16(const float* __restrict__ src,
                             bf16_t* __restrict__ dst, int n) {
    int i = blockIdx.x * blockDim.x + threadIdx.x;
    if (i < n) dst[i] = to_bf16(src[i]);
}

// ---------------------------------------------------------------------------
// NT GEMM: C[M,N] = A[M,K] * B[N,K]^T   (A,B bf16 row-major, K-major)
// MODE 0: Q epilogue  -> bf16 [B,H,S,DH], scaled by 1/sqrt(DH)
// MODE 1: K epilogue  -> bf16 [B,H,S,DH]
// MODE 2: V epilogue  -> bf16 [B,H,DH,S]  (transposed for PV wmma B-operand)
// MODE 3: fp32 row-major [M,N] (final projection, streamed non-temporal)
// Block: 256 threads = 8 waves (4x2), 128x64 tile, 32x32 per wave, K step 32.
// Tiles staged via TDM tensor_load_to_lds, double-buffered on TENSORcnt.
// ---------------------------------------------------------------------------
template <int MODE>
__launch_bounds__(256)
__global__ void gemm_bf16_nt(const bf16_t* __restrict__ A,
                             const bf16_t* __restrict__ Bm,
                             void* __restrict__ Cp,
                             int M, int N, int K) {
    __shared__ bf16_t As[2][128 * 32];   // 16 KB
    __shared__ bf16_t Bs[2][64 * 32];    //  8 KB

    const int tid  = threadIdx.x;
    const int lane = tid & 31;
    const int wv   = tid >> 5;
    const int l15  = lane & 15;
    const int hs   = lane >> 4;          // half-wave select
    const int bm   = blockIdx.y * 128;
    const int bn   = blockIdx.x * 64;
    const int wm   = (wv & 3) * 32;      // wave M offset in tile
    const int wn   = (wv >> 2) * 32;     // wave N offset in tile

    v8f c00 = {}, c01 = {}, c10 = {}, c11 = {};

#if HAVE_TDM
    if (wv == 0) {
        tdm_load_2d((unsigned)(uintptr_t)&As[0][0], A  + (size_t)bm * K,
                    (unsigned)K, (unsigned)(M - bm), (unsigned)K, 32u, 128u);
        tdm_load_2d((unsigned)(uintptr_t)&Bs[0][0], Bm + (size_t)bn * K,
                    (unsigned)K, (unsigned)(N - bn), (unsigned)K, 32u, 64u);
    }
#endif

    for (int k0 = 0; k0 < K; k0 += 32) {
        const int cur = (k0 >> 5) & 1;
#if HAVE_TDM
        if (wv == 0) {
            if (k0 + 32 < K) {
                const int nxt = cur ^ 1;
                tdm_load_2d((unsigned)(uintptr_t)&As[nxt][0],
                            A + (size_t)bm * K + (k0 + 32),
                            (unsigned)(K - k0 - 32), (unsigned)(M - bm), (unsigned)K, 32u, 128u);
                tdm_load_2d((unsigned)(uintptr_t)&Bs[nxt][0],
                            Bm + (size_t)bn * K + (k0 + 32),
                            (unsigned)(K - k0 - 32), (unsigned)(N - bn), (unsigned)K, 32u, 64u);
                __builtin_amdgcn_s_wait_tensorcnt(2);  // retire current buffer's pair
            } else {
                __builtin_amdgcn_s_wait_tensorcnt(0);
            }
        }
        __syncthreads();
#else
        {   // fallback: cooperative VGPR-bounce loads (16B per thread per tile slice)
            const int ldr = tid * 8;
            const int r = ldr >> 5, c = ldr & 31;
            *(bf16x8*)&As[cur][ldr]        = *(const bf16x8*)&A [(size_t)(bm + r) * K + k0 + c];
            *(bf16x8*)&As[cur][ldr + 2048] = *(const bf16x8*)&A [(size_t)(bm + 64 + r) * K + k0 + c];
            *(bf16x8*)&Bs[cur][ldr]        = *(const bf16x8*)&Bm[(size_t)(bn + r) * K + k0 + c];
        }
        __syncthreads();
#endif
        // A fragments (16x32 bf16, lane packing: half0 K{0..7,16..23}, half1 K{8..15,24..31})
        const bf16_t* a0p = &As[cur][(wm + l15) * 32];
        const bf16_t* a1p = &As[cur][(wm + 16 + l15) * 32];
        v16bf a0 = combine16(*(const bf16x8*)&a0p[hs * 8], *(const bf16x8*)&a0p[16 + hs * 8]);
        v16bf a1 = combine16(*(const bf16x8*)&a1p[hs * 8], *(const bf16x8*)&a1p[16 + hs * 8]);
        // B fragments (32x16: half0 K{0..7,16..23}, half1 K{8..15,24..31}, column = lane&15)
        const bf16_t* b0p = &Bs[cur][(wn + l15) * 32];
        const bf16_t* b1p = &Bs[cur][(wn + 16 + l15) * 32];
        v16bf b0 = combine16(*(const bf16x8*)&b0p[hs * 8], *(const bf16x8*)&b0p[16 + hs * 8]);
        v16bf b1 = combine16(*(const bf16x8*)&b1p[hs * 8], *(const bf16x8*)&b1p[16 + hs * 8]);

        c00 = wmma_bf16(a0, b0, c00);
        c01 = wmma_bf16(a0, b1, c01);
        c10 = wmma_bf16(a1, b0, c10);
        c11 = wmma_bf16(a1, b1, c11);
        __syncthreads();
    }

    // Epilogue. C fragment: col = lane&15, row = vgpr + 8*(lane>=16).
#pragma unroll
    for (int i = 0; i < 2; ++i) {
#pragma unroll
        for (int j = 0; j < 2; ++j) {
            v8f c = (i == 0) ? (j == 0 ? c00 : c01) : (j == 0 ? c10 : c11);
#pragma unroll
            for (int v = 0; v < 8; ++v) {
                const int rg = bm + wm + i * 16 + v + hs * 8;
                const int cg = bn + wn + j * 16 + l15;
                const float val = c[v];
                if (MODE == 3) {
                    __builtin_nontemporal_store(val, &((float*)Cp)[(size_t)rg * N + cg]);
                } else {
                    const int bb = rg >> 11, ss = rg & (kS - 1);
                    const int hd = cg >> 6,  dh = cg & (kDH - 1);
                    bf16_t* D = (bf16_t*)Cp;
                    if (MODE == 0)
                        D[((size_t)(bb * kH + hd) * kS + ss) * kDH + dh] = to_bf16(val * 0.125f);
                    else if (MODE == 1)
                        D[((size_t)(bb * kH + hd) * kS + ss) * kDH + dh] = to_bf16(val);
                    else
                        D[((size_t)(bb * kH + hd) * kDH + dh) * kS + ss] = to_bf16(val);
                }
            }
        }
    }
}

// ---------------------------------------------------------------------------
// Flash attention: one wave per 16-query tile, 32-key steps, online softmax.
// Q,K: bf16 [B,H,S,64]; V: bf16 [B,H,64,S]; O: bf16 [B*S, D] row-major.
// 1/sqrt(DH) already folded into Q.
// ---------------------------------------------------------------------------
__launch_bounds__(128)
__global__ void flash_attn(const bf16_t* __restrict__ Q,
                           const bf16_t* __restrict__ Kh,
                           const bf16_t* __restrict__ Vt,
                           bf16_t* __restrict__ O) {
    __shared__ bf16_t Plds[4][16 * 32];   // per-wave P tile (C-layout -> A-layout bounce)

    const int lane = threadIdx.x & 31;
    const int wv   = threadIdx.x >> 5;
    const int l15  = lane & 15;
    const int hs   = lane >> 4;

    const int qtiles = kS / 16;           // 128
    const int g   = blockIdx.x * 4 + wv;
    const int bh  = g / qtiles;
    const int q0  = (g % qtiles) * 16;

    const bf16_t* Qb = Q  + (size_t)bh * kS * kDH;
    const bf16_t* Kb = Kh + (size_t)bh * kS * kDH;
    const bf16_t* Vb = Vt + (size_t)bh * kDH * kS;

    // Q A-fragments for K-dim 64 (two 16x32 fragments)
    const bf16_t* qp = &Qb[(size_t)(q0 + l15) * kDH];
    v16bf qa0 = combine16(*(const bf16x8*)&qp[hs * 8],      *(const bf16x8*)&qp[16 + hs * 8]);
    v16bf qa1 = combine16(*(const bf16x8*)&qp[32 + hs * 8], *(const bf16x8*)&qp[48 + hs * 8]);

    v8f acc0 = {}, acc1 = {}, acc2 = {}, acc3 = {};
    float mrow[8], lrow[8];
#pragma unroll
    for (int v = 0; v < 8; ++v) { mrow[v] = -3.0e38f; lrow[v] = 0.0f; }

    const int nkt = (q0 + 16 + 31) >> 5;  // causal: only key tiles <= query tile
    for (int kt = 0; kt < nkt; ++kt) {
        const int kk0 = kt * 32;

        // ---- scores: S = Q(16x64) @ K^T(64x32) as two 16x16 fragments ----
        v8f s01[2];
#pragma unroll
        for (int half = 0; half < 2; ++half) {
            const int kk = kk0 + half * 16;
            const bf16_t* kp = &Kb[(size_t)(kk + l15) * kDH];
            v16bf kb0 = combine16(*(const bf16x8*)&kp[hs * 8],      *(const bf16x8*)&kp[16 + hs * 8]);
            v16bf kb1 = combine16(*(const bf16x8*)&kp[32 + hs * 8], *(const bf16x8*)&kp[48 + hs * 8]);
            v8f s = {};
            s = wmma_bf16(qa0, kb0, s);
            s = wmma_bf16(qa1, kb1, s);
            s01[half] = s;
        }

        // ---- causal mask + row max (reduce across 16 lanes of half-wave) ----
        float mnew[8], alpha[8];
#pragma unroll
        for (int v = 0; v < 8; ++v) {
            const int rg = q0 + v + hs * 8;
            const float s0 = ((kk0 + l15)      <= rg) ? s01[0][v] : -3.0e38f;
            const float s1 = ((kk0 + 16 + l15) <= rg) ? s01[1][v] : -3.0e38f;
            s01[0][v] = s0; s01[1][v] = s1;
            float r = fmaxf(s0, s1);
            r = fmaxf(r, __shfl_xor(r, 1, 32));
            r = fmaxf(r, __shfl_xor(r, 2, 32));
            r = fmaxf(r, __shfl_xor(r, 4, 32));
            r = fmaxf(r, __shfl_xor(r, 8, 32));
            mnew[v] = fmaxf(mrow[v], r);
        }

        // ---- P = exp(S - m_new), row sums, online update of l ----
#pragma unroll
        for (int v = 0; v < 8; ++v) {
            const float p0 = __expf(s01[0][v] - mnew[v]);
            const float p1 = __expf(s01[1][v] - mnew[v]);
            s01[0][v] = p0; s01[1][v] = p1;
            float r = p0 + p1;
            r += __shfl_xor(r, 1, 32);
            r += __shfl_xor(r, 2, 32);
            r += __shfl_xor(r, 4, 32);
            r += __shfl_xor(r, 8, 32);
            const float al = __expf(mrow[v] - mnew[v]);
            lrow[v] = lrow[v] * al + r;
            mrow[v] = mnew[v];
            alpha[v] = al;
        }

        // ---- rescale accumulators ----
#pragma unroll
        for (int v = 0; v < 8; ++v) {
            const float al = alpha[v];
            acc0[v] *= al; acc1[v] *= al; acc2[v] *= al; acc3[v] *= al;
        }

        // ---- P: C-layout -> A-layout via per-wave LDS tile ----
        bf16_t* pl = Plds[wv];
#pragma unroll
        for (int v = 0; v < 8; ++v) {
            const int row = v + hs * 8;
            pl[row * 32 + l15]      = to_bf16(s01[0][v]);
            pl[row * 32 + 16 + l15] = to_bf16(s01[1][v]);
        }
        asm volatile("s_wait_dscnt 0" ::: "memory");   // wave-local LDS fence
        const bf16_t* pp = &pl[l15 * 32];
        v16bf pa = combine16(*(const bf16x8*)&pp[hs * 8], *(const bf16x8*)&pp[16 + hs * 8]);

        // ---- acc += P(16x32) @ V(32x64), V^T stored so k is unit-stride ----
#pragma unroll
        for (int d = 0; d < 4; ++d) {
            const bf16_t* vp = &Vb[(size_t)(d * 16 + l15) * kS + kk0];
            v16bf vb = combine16(*(const bf16x8*)&vp[hs * 8], *(const bf16x8*)&vp[16 + hs * 8]);
            v8f& ac = d == 0 ? acc0 : d == 1 ? acc1 : d == 2 ? acc2 : acc3;
            ac = wmma_bf16(pa, vb, ac);
        }
    }

    // ---- epilogue: O[b*S+row, head*64 + d*16 + col] = acc / l ----
    const int b    = bh >> 4;
    const int head = bh & 15;
#pragma unroll
    for (int v = 0; v < 8; ++v) {
        const int row = q0 + v + hs * 8;
        const float inv = 1.0f / lrow[v];
        const size_t base = ((size_t)(b * kS + row)) * kD + head * kDH + l15;
        O[base + 0]  = to_bf16(acc0[v] * inv);
        O[base + 16] = to_bf16(acc1[v] * inv);
        O[base + 32] = to_bf16(acc2[v] * inv);
        O[base + 48] = to_bf16(acc3[v] * inv);
    }
}

// ---------------------------------------------------------------------------
// Launch
// ---------------------------------------------------------------------------
extern "C" void kernel_launch(void* const* d_in, const int* in_sizes, int n_in,
                              void* d_out, int out_size, void* d_ws, size_t ws_size,
                              hipStream_t stream) {
    const float* x  = (const float*)d_in[0];
    // d_in[1] = causal mask (applied analytically)
    const float* Wq = (const float*)d_in[2];
    const float* Wk = (const float*)d_in[3];
    const float* Wv = (const float*)d_in[4];
    const float* Wo = (const float*)d_in[5];

    char* ws = (char*)d_ws;
    size_t off = 0;
    bf16_t* xb  = (bf16_t*)(ws + off); off += (size_t)kM * kD * 2;
    bf16_t* wqb = (bf16_t*)(ws + off); off += (size_t)kD * kD * 2;
    bf16_t* wkb = (bf16_t*)(ws + off); off += (size_t)kD * kD * 2;
    bf16_t* wvb = (bf16_t*)(ws + off); off += (size_t)kD * kD * 2;
    bf16_t* wob = (bf16_t*)(ws + off); off += (size_t)kD * kD * 2;
    bf16_t* Qb  = (bf16_t*)(ws + off); off += (size_t)kM * kD * 2;
    bf16_t* Kb  = (bf16_t*)(ws + off); off += (size_t)kM * kD * 2;
    bf16_t* Vt  = (bf16_t*)(ws + off); off += (size_t)kM * kD * 2;
    bf16_t* Ob  = (bf16_t*)(ws + off); off += (size_t)kM * kD * 2;

    const int nX = kM * kD;
    const int nW = kD * kD;
    cvt_f32_bf16<<<(nX + 255) / 256, 256, 0, stream>>>(x,  xb,  nX);
    cvt_f32_bf16<<<(nW + 255) / 256, 256, 0, stream>>>(Wq, wqb, nW);
    cvt_f32_bf16<<<(nW + 255) / 256, 256, 0, stream>>>(Wk, wkb, nW);
    cvt_f32_bf16<<<(nW + 255) / 256, 256, 0, stream>>>(Wv, wvb, nW);
    cvt_f32_bf16<<<(nW + 255) / 256, 256, 0, stream>>>(Wo, wob, nW);

    dim3 gg(kD / 64, kM / 128);
    gemm_bf16_nt<0><<<gg, 256, 0, stream>>>(xb, wqb, Qb, kM, kD, kD);
    gemm_bf16_nt<1><<<gg, 256, 0, stream>>>(xb, wkb, Kb, kM, kD, kD);
    gemm_bf16_nt<2><<<gg, 256, 0, stream>>>(xb, wvb, Vt, kM, kD, kD);

    flash_attn<<<(kB * kH * (kS / 16)) / 4, 128, 0, stream>>>(Qb, Kb, Vt, Ob);

    gemm_bf16_nt<3><<<gg, 256, 0, stream>>>(Ob, wob, d_out, kM, kD, kD);
}
